// AdvancedFeatureGNN_16329465660175
// MI455X (gfx1250) — matrix operand order repaired and checked by reading, not compiled, not run on hardware
//
#include <hip/hip_runtime.h>
#include <hip/hip_bf16.h>

typedef __attribute__((ext_vector_type(16))) _Float16 v16h;
typedef __attribute__((ext_vector_type(8)))  float    v8f;

#define HDIM 128
#define BN_EPS 1e-5f

// ---------------- utility kernels ----------------

__global__ void zero4_kernel(float4* p, int n4) {
  int i = blockIdx.x * blockDim.x + threadIdx.x;
  if (i < n4) p[i] = make_float4(0.f, 0.f, 0.f, 0.f);
}

__global__ void init_deg_kernel(float* deg, int n) {
  int i = blockIdx.x * blockDim.x + threadIdx.x;
  if (i < n) deg[i] = 1.0f;  // self loop
}

__global__ void edge_deg_kernel(const int* __restrict__ dst, float* deg, int e) {
  int i = blockIdx.x * blockDim.x + threadIdx.x;
  if (i < e) atomicAdd(&deg[dst[i]], 1.0f);
}

__global__ void rsqrt_kernel(float* deg, int n) {
  int i = blockIdx.x * blockDim.x + threadIdx.x;
  if (i < n) deg[i] = rsqrtf(deg[i]);   // deg buffer becomes dis
}

__global__ void enorm_kernel(const int* __restrict__ src, const int* __restrict__ dst,
                             const float* __restrict__ dis, float* enorm, int e) {
  int i = blockIdx.x * blockDim.x + threadIdx.x;
  if (i < e) enorm[i] = dis[src[i]] * dis[dst[i]];
}

// ---------------- WMMA GEMM: C[M,128] = A[M,128] @ W[128,128], f16 inputs / f32 acc -------
// Block: 256 threads = 8 waves; each wave computes a 16-row x 128-col strip.
// W is staged transposed (lw[n*128+k]) as f16 in LDS so B fragments are 2x ds_load_b128.
// All 16 A-row float4 loads are issued up front so global latency overlaps.

__global__ __launch_bounds__(256) void gemm_wmma_kernel(const float* __restrict__ A,
                                                        const float* __restrict__ W,
                                                        float* __restrict__ C, int M) {
  __shared__ __align__(32) _Float16 lw[HDIM * HDIM];
  int t = threadIdx.x;

  const int wave   = t >> 5;
  const int lane   = t & 31;
  const int khalf  = lane >> 4;       // 0/1: which K-half this lane holds
  const int l15    = lane & 15;
  const int rowBase = blockIdx.x * 128 + wave * 16;

  // A-fragment source row for this lane (ISA: 16-bit A, lanes 0-15 -> M=0..15, 16-31 repeat M)
  const int arow_i = rowBase + l15;
  const int rowc   = arow_i < M ? arow_i : (M - 1);
  const float sc   = arow_i < M ? 1.0f : 0.0f;   // select, no divergent branch: EXEC stays full
  const float* arow = A + (size_t)rowc * HDIM;

  // ---- issue ALL A loads first (overlaps with the W->LDS staging below) ----
  float4 fa[16];
#pragma unroll
  for (int q = 0; q < 4; ++q) {
    const int kk = q * 32;
    fa[q * 4 + 0] = *(const float4*)(arow + kk + khalf * 16);
    fa[q * 4 + 1] = *(const float4*)(arow + kk + khalf * 16 + 4);
    fa[q * 4 + 2] = *(const float4*)(arow + kk + khalf * 16 + 8);
    fa[q * 4 + 3] = *(const float4*)(arow + kk + khalf * 16 + 12);
  }

  // ---- stage W transposed as f16 into LDS ----
  for (int i = t; i < HDIM * HDIM; i += 256) {
    int k = i >> 7, n = i & (HDIM - 1);
    lw[n * HDIM + k] = (_Float16)W[i];
  }

  // ---- convert A to four 16-half fragments while LDS staging drains ----
  v16h afr[4];
#pragma unroll
  for (int q = 0; q < 4; ++q) {
    float av[16] = {fa[q*4+0].x, fa[q*4+0].y, fa[q*4+0].z, fa[q*4+0].w,
                    fa[q*4+1].x, fa[q*4+1].y, fa[q*4+1].z, fa[q*4+1].w,
                    fa[q*4+2].x, fa[q*4+2].y, fa[q*4+2].z, fa[q*4+2].w,
                    fa[q*4+3].x, fa[q*4+3].y, fa[q*4+3].z, fa[q*4+3].w};
#pragma unroll
    for (int j = 0; j < 16; ++j) afr[q][j] = (_Float16)(av[j] * sc);
  }

  __syncthreads();

  v8f acc[8];
#pragma unroll
  for (int nt = 0; nt < 8; ++nt) {
    v8f z = {0.f, 0.f, 0.f, 0.f, 0.f, 0.f, 0.f, 0.f};
    acc[nt] = z;
  }

#pragma unroll
  for (int q = 0; q < 4; ++q) {
    // B fragment order used here: lane = (K-half)*16 | n; elems 0..15 -> consecutive k, same as A
    const int kb = q * 32 + khalf * 16;
#pragma unroll
    for (int nt = 0; nt < 8; ++nt) {
      const int ncol = nt * 16 + l15;
      v16h b = *(const v16h*)(&lw[ncol * HDIM + kb]);
      acc[nt] = __builtin_amdgcn_wmma_f32_16x16x32_f16(false, afr[q], false, b,
                                                       (short)0, acc[nt], false, false);
    }
  }

  // C/D layout: VGPR r, lanes 0-15 -> M=r, lanes 16-31 -> M=8+r; N = lane&15
  const int srow = rowBase + khalf * 8;
#pragma unroll
  for (int nt = 0; nt < 8; ++nt) {
#pragma unroll
    for (int r = 0; r < 8; ++r) {
      int rr = srow + r;
      if (rr < M) C[(size_t)rr * HDIM + nt * 16 + l15] = acc[nt][r];
    }
  }
}

// ---------------- message passing ----------------

// out = hW * dis[node]^2 (self-loop term), float4-vectorized
__global__ void self_init_kernel(const float4* __restrict__ hW, const float* __restrict__ dis,
                                 float4* __restrict__ out, int total4) {
  int i = blockIdx.x * blockDim.x + threadIdx.x;
  if (i >= total4) return;
  int node = i >> 5;                      // 32 float4 per 128-feature row
  float d = dis[node];
  float s = d * d;
  float4 v = hW[i];
  out[i] = make_float4(v.x * s, v.y * s, v.z * s, v.w * s);
}

// one wave32 per edge; lane handles 4 consecutive features (float4 gather, 4 f32 atomics)
__global__ void scatter_kernel(const float* __restrict__ hW, const int* __restrict__ src,
                               const int* __restrict__ dst, const float* __restrict__ enorm,
                               float* __restrict__ out, int e) {
  int gid = blockIdx.x * blockDim.x + threadIdx.x;
  int lane = gid & 31;
  int ed = gid >> 5;
  if (ed >= e) return;
  int s = src[ed], d = dst[ed];
  float nv = enorm[ed];
  float4 v = *(const float4*)(hW + (size_t)s * HDIM + lane * 4);
  float* o = out + (size_t)d * HDIM + lane * 4;
  atomicAdd(o + 0, nv * v.x);
  atomicAdd(o + 1, nv * v.y);
  atomicAdd(o + 2, nv * v.z);
  atomicAdd(o + 3, nv * v.w);
}

// fused bias + BN + ReLU, float4-vectorized (in place)
__global__ void bn_relu_kernel(float4* __restrict__ agg, const float4* __restrict__ bias,
                               const float4* __restrict__ g, const float4* __restrict__ be,
                               const float4* __restrict__ m, const float4* __restrict__ v,
                               int total4) {
  int i = blockIdx.x * blockDim.x + threadIdx.x;
  if (i >= total4) return;
  int f4 = i & 31;                        // feature-group index (128/4)
  float4 a = agg[i], bb = bias[f4], gg = g[f4], ee = be[f4], mm = m[f4], vv = v[f4];
  float4 r;
  r.x = fmaxf((a.x + bb.x - mm.x) * gg.x * rsqrtf(vv.x + BN_EPS) + ee.x, 0.f);
  r.y = fmaxf((a.y + bb.y - mm.y) * gg.y * rsqrtf(vv.y + BN_EPS) + ee.y, 0.f);
  r.z = fmaxf((a.z + bb.z - mm.z) * gg.z * rsqrtf(vv.z + BN_EPS) + ee.z, 0.f);
  r.w = fmaxf((a.w + bb.w - mm.w) * gg.w * rsqrtf(vv.w + BN_EPS) + ee.w, 0.f);
  agg[i] = r;
}

// ---------------- pooling ----------------

__global__ void count_kernel(const int* __restrict__ batch, float* counts, int n) {
  int i = blockIdx.x * blockDim.x + threadIdx.x;
  if (i < n) atomicAdd(&counts[batch[i]], 1.0f);
}

// fcat layout per graph: [0:128)=mean, [128:256)=max, [256:384)=sum, [384:512)=gf
// float4 gather of h; post-ReLU values >= 0 so int-bit atomicMax is order-correct,
// and the zero-init matches where(counts>0, max, 0).
__global__ void pool_kernel(const float4* __restrict__ h, const int* __restrict__ batch,
                            float* __restrict__ fcat, int total4) {
  int i = blockIdx.x * blockDim.x + threadIdx.x;
  if (i >= total4) return;
  int node = i >> 5, f = (i & 31) * 4;
  int g = batch[node];
  float4 v = h[i];
  float* s = &fcat[(size_t)g * 512 + 256 + f];
  atomicAdd(s + 0, v.x);
  atomicAdd(s + 1, v.y);
  atomicAdd(s + 2, v.z);
  atomicAdd(s + 3, v.w);
  int* mx = (int*)&fcat[(size_t)g * 512 + 128 + f];
  atomicMax(mx + 0, __float_as_int(v.x));
  atomicMax(mx + 1, __float_as_int(v.y));
  atomicMax(mx + 2, __float_as_int(v.z));
  atomicMax(mx + 3, __float_as_int(v.w));
}

__global__ void mean_kernel(float* __restrict__ fcat, const float* __restrict__ counts, int G) {
  int i = blockIdx.x * blockDim.x + threadIdx.x;
  if (i >= G * HDIM) return;
  int g = i >> 7, f = i & (HDIM - 1);
  fcat[(size_t)g * 512 + f] = fcat[(size_t)g * 512 + 256 + f] / fmaxf(counts[g], 1.0f);
}

// ---------------- small dense layers (FC [+BN+ReLU]) ----------------

__global__ void fc_kernel(const float* __restrict__ in, int in_stride,
                          const float* __restrict__ W, const float* __restrict__ bias,
                          const float* __restrict__ g, const float* __restrict__ be,
                          const float* __restrict__ m, const float* __restrict__ v,
                          int bn_relu, float* __restrict__ out, int out_stride, int out_off,
                          int R, int K, int C) {
  int idx = blockIdx.x * blockDim.x + threadIdx.x;
  if (idx >= R * C) return;
  int r = idx / C, c = idx % C;
  const float* ip = in + (size_t)r * in_stride;
  float acc = bias[c];
  for (int k = 0; k < K; ++k) acc += ip[k] * W[k * C + c];
  if (bn_relu) {
    acc = (acc - m[c]) * g[c] * rsqrtf(v[c] + BN_EPS) + be[c];
    acc = fmaxf(acc, 0.0f);
  }
  out[(size_t)r * out_stride + out_off + c] = acc;
}

// ---------------- launch ----------------

static inline int nblk(long n, int b) { return (int)((n + b - 1) / b); }

extern "C" void kernel_launch(void* const* d_in, const int* in_sizes, int n_in,
                              void* d_out, int out_size, void* d_ws, size_t ws_size,
                              hipStream_t stream) {
  const float* x      = (const float*)d_in[0];
  const float* gfe    = (const float*)d_in[1];
  const int*   ei     = (const int*)d_in[2];
  const int*   batch  = (const int*)d_in[3];
  const float* conv_W = (const float*)d_in[4];
  const float* conv_b = (const float*)d_in[5];
  const float* bn_g   = (const float*)d_in[6];
  const float* bn_b   = (const float*)d_in[7];
  const float* bn_m   = (const float*)d_in[8];
  const float* bn_v   = (const float*)d_in[9];
  const float* gm_W1  = (const float*)d_in[10];
  const float* gm_b1  = (const float*)d_in[11];
  const float* gm_g1  = (const float*)d_in[12];
  const float* gm_be1 = (const float*)d_in[13];
  const float* gm_m1  = (const float*)d_in[14];
  const float* gm_v1  = (const float*)d_in[15];
  const float* gm_W2  = (const float*)d_in[16];
  const float* gm_b2  = (const float*)d_in[17];
  const float* gm_g2  = (const float*)d_in[18];
  const float* gm_be2 = (const float*)d_in[19];
  const float* gm_m2  = (const float*)d_in[20];
  const float* gm_v2  = (const float*)d_in[21];
  const float* fu_W1  = (const float*)d_in[22];
  const float* fu_b1  = (const float*)d_in[23];
  const float* fu_g1  = (const float*)d_in[24];
  const float* fu_be1 = (const float*)d_in[25];
  const float* fu_m1  = (const float*)d_in[26];
  const float* fu_v1  = (const float*)d_in[27];
  const float* fu_W2  = (const float*)d_in[28];
  const float* fu_b2  = (const float*)d_in[29];
  const float* fu_g2  = (const float*)d_in[30];
  const float* fu_be2 = (const float*)d_in[31];
  const float* fu_m2  = (const float*)d_in[32];
  const float* fu_v2  = (const float*)d_in[33];
  const float* fu_W3  = (const float*)d_in[34];
  const float* fu_b3  = (const float*)d_in[35];

  const int Nn = in_sizes[0] / HDIM;   // 50000
  const int Ee = in_sizes[2] / 2;      // 800000
  const int G  = in_sizes[1] / 64;     // 512
  const int* srcp = ei;
  const int* dstp = ei + Ee;
  const size_t nh  = (size_t)Nn * HDIM;
  const int    nh4 = (int)(nh / 4);

  // workspace carve-out (floats)
  float* ws = (float*)d_ws;
  size_t off = 0;
  auto take = [&](size_t n) { float* p = ws + off; off += (n + 255) & ~(size_t)255; return p; };
  float* dis    = take(Nn);           // deg -> dis (in place)
  float* enorm  = take(Ee);
  float* P0     = take(nh);
  float* P1     = take(nh);
  float* P2     = take(nh);
  float* counts = take(G);
  float* fcat   = take((size_t)G * 512);
  float* tmp1   = take((size_t)G * 128);
  float* u1     = take((size_t)G * 256);
  float* u2     = take((size_t)G * 128);
  (void)ws_size; (void)n_in; (void)out_size;

  // --- normalization coefficients ---
  init_deg_kernel<<<nblk(Nn, 256), 256, 0, stream>>>(dis, Nn);
  edge_deg_kernel<<<nblk(Ee, 256), 256, 0, stream>>>(dstp, dis, Ee);
  rsqrt_kernel<<<nblk(Nn, 256), 256, 0, stream>>>(dis, Nn);
  enorm_kernel<<<nblk(Ee, 256), 256, 0, stream>>>(srcp, dstp, dis, enorm, Ee);

  // --- 4 GCN layers, rotating 3 node-sized buffers ---
  float* bufs[3] = {P0, P1, P2};
  const float* hin = x;
  int in_idx = -1;
  for (int l = 0; l < 4; ++l) {
    int a = -1, b = -1;
    for (int j = 0; j < 3; ++j) {
      if (j == in_idx) continue;
      if (a < 0) a = j; else if (b < 0) b = j;
    }
    float* hW  = bufs[a];
    float* agg = bufs[b];

    gemm_wmma_kernel<<<nblk(Nn, 128), 256, 0, stream>>>(hin, conv_W + (size_t)l * HDIM * HDIM, hW, Nn);
    self_init_kernel<<<nblk(nh4, 256), 256, 0, stream>>>((const float4*)hW, dis, (float4*)agg, nh4);
    scatter_kernel<<<nblk((long)Ee * 32, 256), 256, 0, stream>>>(hW, srcp, dstp, enorm, agg, Ee);
    bn_relu_kernel<<<nblk(nh4, 256), 256, 0, stream>>>(
        (float4*)agg, (const float4*)(conv_b + l * HDIM), (const float4*)(bn_g + l * HDIM),
        (const float4*)(bn_b + l * HDIM), (const float4*)(bn_m + l * HDIM),
        (const float4*)(bn_v + l * HDIM), nh4);

    hin = agg;
    in_idx = b;
  }

  // --- pooling into fcat ---
  zero4_kernel<<<nblk((long)G * 128, 256), 256, 0, stream>>>((float4*)fcat, G * 128);
  zero4_kernel<<<nblk(G / 4, 256), 256, 0, stream>>>((float4*)counts, G / 4);
  count_kernel<<<nblk(Nn, 256), 256, 0, stream>>>(batch, counts, Nn);
  pool_kernel<<<nblk(nh4, 256), 256, 0, stream>>>((const float4*)hin, batch, fcat, nh4);
  mean_kernel<<<nblk((long)G * HDIM, 256), 256, 0, stream>>>(fcat, counts, G);

  // --- graph-feature MLP -> fcat[:, 384:512) ---
  fc_kernel<<<nblk((long)G * 128, 128), 128, 0, stream>>>(
      gfe, 64, gm_W1, gm_b1, gm_g1, gm_be1, gm_m1, gm_v1, 1, tmp1, 128, 0, G, 64, 128);
  fc_kernel<<<nblk((long)G * 128, 128), 128, 0, stream>>>(
      tmp1, 128, gm_W2, gm_b2, gm_g2, gm_be2, gm_m2, gm_v2, 1, fcat, 512, 384, G, 128, 128);

  // --- fusion MLP ---
  fc_kernel<<<nblk((long)G * 256, 128), 128, 0, stream>>>(
      fcat, 512, fu_W1, fu_b1, fu_g1, fu_be1, fu_m1, fu_v1, 1, u1, 256, 0, G, 512, 256);
  fc_kernel<<<nblk((long)G * 128, 128), 128, 0, stream>>>(
      u1, 256, fu_W2, fu_b2, fu_g2, fu_be2, fu_m2, fu_v2, 1, u2, 128, 0, G, 256, 128);
  fc_kernel<<<nblk(G, 128), 128, 0, stream>>>(
      u2, 128, fu_W3, fu_b3, nullptr, nullptr, nullptr, nullptr, 0, (float*)d_out, 1, 0, G, 128, 1);
}
